// LSAMamba3Block_46050639348062
// MI455X (gfx1250) — matrix operand order repaired and checked by reading, not compiled, not run on hardware
//
#include <hip/hip_runtime.h>
#include <hip/hip_bf16.h>
#include <math.h>

#define T_LEN 2048
#define D_MODEL 2048
#define N_HEADS 16
#define DH 64
#define D_LAT 512
#define D_STATE 128
#define D_FF 8192
#define NH_M 8
#define HD_M 64

typedef __attribute__((ext_vector_type(16))) __bf16 v16bf;
typedef __attribute__((ext_vector_type(8)))  __bf16 v8bf;
typedef __attribute__((ext_vector_type(8)))  float  v8f;

static __device__ inline v8f zero8() {
    v8f z;
#pragma unroll
    for (int i = 0; i < 8; ++i) z[i] = 0.0f;
    return z;
}

static __device__ inline v16bf mk16(v8bf a, v8bf b) {
    v16bf r;
#pragma unroll
    for (int i = 0; i < 8; ++i) { r[i] = a[i]; r[8 + i] = b[i]; }
    return r;
}

// ---------------------------------------------------------------------------
// RMSNorm: one block per row
// ---------------------------------------------------------------------------
__global__ __launch_bounds__(256) void rmsnorm_kernel(const float* __restrict__ x,
                                                      const float* __restrict__ w,
                                                      float* __restrict__ out, int D) {
    __shared__ float red[256];
    const int row = blockIdx.x;
    const float* xr = x + (size_t)row * D;
    float s = 0.f;
    for (int i = threadIdx.x; i < D; i += 256) { float v = xr[i]; s += v * v; }
    red[threadIdx.x] = s;
    __syncthreads();
    for (int off = 128; off > 0; off >>= 1) {
        if (threadIdx.x < off) red[threadIdx.x] += red[threadIdx.x + off];
        __syncthreads();
    }
    const float inv = rsqrtf(red[0] / (float)D + 1e-6f);
    float* orow = out + (size_t)row * D;
    for (int i = threadIdx.x; i < D; i += 256) orow[i] = xr[i] * inv * w[i];
}

// ---------------------------------------------------------------------------
// Generic WMMA bf16 GEMM: C[M,N] = A[M,K] @ B[K,N] (+R). Tiles 128x128, K step 32.
// 256 threads = 8 waves, wave grid 2(M)x4(N): each wave 4 Mtiles x 2 Ntiles.
// Software-pipelined: next k-tile is fetched into VGPRs while WMMAs run on the
// current LDS tile, so global loads stay in flight behind compute.
// ---------------------------------------------------------------------------
__global__ __launch_bounds__(256) void gemm_bf16_kernel(const float* __restrict__ A,
                                                        const float* __restrict__ B,
                                                        const float* __restrict__ R,
                                                        float* __restrict__ C,
                                                        int M, int N, int K) {
    __shared__ __bf16 aT[128 * 32];   // [m][k] row major
    __shared__ __bf16 bT[128 * 32];   // [n][k] (B transposed)

    const int tid  = threadIdx.x;
    const int wave = tid >> 5;
    const int lane = tid & 31;
    const int wm   = wave >> 2;          // 0..1
    const int wn   = wave & 3;           // 0..3
    const int l15  = lane & 15;
    const int loA  = (lane < 16) ? 0 : 8;    // A K-half select
    const int koB  = (lane < 16) ? 0 : 8;    // B K-half select
    const int hi8  = (lane < 16) ? 0 : 8;    // C row-half select

    const int m0 = blockIdx.y * 128;
    const int n0 = blockIdx.x * 128;

    // per-thread staging coordinates
    const int amRow = tid >> 3;               // A: row within tile, 4 rows apart per it? no: idx-based below
    (void)amRow;

    v8f acc[4][2];
#pragma unroll
    for (int i = 0; i < 4; ++i)
#pragma unroll
        for (int j = 0; j < 2; ++j) acc[i][j] = zero8();

    float4 aReg[4], bReg[4];

    auto fetch = [&](int k0) {
#pragma unroll
        for (int it = 0; it < 4; ++it) {
            int idx = tid + it * 256;          // 1024 quads, A tile 128x32
            int m = idx >> 3;
            int k = (idx & 7) * 4;
            aReg[it] = *reinterpret_cast<const float4*>(&A[(size_t)(m0 + m) * K + k0 + k]);
        }
#pragma unroll
        for (int it = 0; it < 4; ++it) {
            int idx = tid + it * 256;          // B tile 32x128, read along N
            int k = idx >> 5;
            int n = (idx & 31) * 4;
            bReg[it] = *reinterpret_cast<const float4*>(&B[(size_t)(k0 + k) * N + n0 + n]);
        }
    };
    auto stage = [&]() {
#pragma unroll
        for (int it = 0; it < 4; ++it) {
            int idx = tid + it * 256;
            int m = idx >> 3;
            int k = (idx & 7) * 4;
            __bf16* d = &aT[m * 32 + k];
            d[0] = (__bf16)aReg[it].x; d[1] = (__bf16)aReg[it].y;
            d[2] = (__bf16)aReg[it].z; d[3] = (__bf16)aReg[it].w;
        }
#pragma unroll
        for (int it = 0; it < 4; ++it) {
            int idx = tid + it * 256;
            int k = idx >> 5;
            int n = (idx & 31) * 4;
            bT[(n + 0) * 32 + k] = (__bf16)bReg[it].x;
            bT[(n + 1) * 32 + k] = (__bf16)bReg[it].y;
            bT[(n + 2) * 32 + k] = (__bf16)bReg[it].z;
            bT[(n + 3) * 32 + k] = (__bf16)bReg[it].w;
        }
    };

    fetch(0);
    for (int k0 = 0; k0 < K; k0 += 32) {
        stage();
        __syncthreads();
        if (k0 + 32 < K) fetch(k0 + 32);       // in flight behind the WMMAs below

        v16bf aOp[4];
#pragma unroll
        for (int i = 0; i < 4; ++i) {
            int row = wm * 64 + i * 16 + l15;
            v8bf lo = *reinterpret_cast<const v8bf*>(&aT[row * 32 + loA]);
            v8bf hi = *reinterpret_cast<const v8bf*>(&aT[row * 32 + 16 + loA]);
            aOp[i] = mk16(lo, hi);
        }
        v16bf bOp[2];
#pragma unroll
        for (int j = 0; j < 2; ++j) {
            int n = wn * 32 + j * 16 + l15;
            v8bf lo = *reinterpret_cast<const v8bf*>(&bT[n * 32 + koB]);
            v8bf hi = *reinterpret_cast<const v8bf*>(&bT[n * 32 + koB + 16]);
            bOp[j] = mk16(lo, hi);
        }
#pragma unroll
        for (int i = 0; i < 4; ++i)
#pragma unroll
            for (int j = 0; j < 2; ++j)
                acc[i][j] = __builtin_amdgcn_wmma_f32_16x16x32_bf16(
                    false, aOp[i], false, bOp[j], (short)0, acc[i][j], false, false);
        __syncthreads();
    }

    // epilogue
#pragma unroll
    for (int i = 0; i < 4; ++i) {
#pragma unroll
        for (int j = 0; j < 2; ++j) {
            int col = n0 + wn * 32 + j * 16 + l15;
#pragma unroll
            for (int r = 0; r < 8; ++r) {
                int row = m0 + wm * 64 + i * 16 + r + hi8;
                float v = acc[i][j][r];
                if (R) v += R[(size_t)row * N + col];
                C[(size_t)row * N + col] = v;
            }
        }
    }
}

// ---------------------------------------------------------------------------
// dt / decay: dt[t][h] = softplus(c_kv[t]·W_dt[:,h] + bias), dec = exp(-exp(A_log)*dt)
// ---------------------------------------------------------------------------
__global__ __launch_bounds__(256) void dt_kernel(const float* __restrict__ c_kv,
                                                 const float* __restrict__ W_dt,
                                                 const float* __restrict__ dt_bias,
                                                 const float* __restrict__ A_log,
                                                 float* __restrict__ dt,
                                                 float* __restrict__ dec) {
    int id = blockIdx.x * 256 + threadIdx.x;   // T_LEN * NH_M
    if (id >= T_LEN * NH_M) return;
    int t = id >> 3, h = id & 7;
    const float* c = c_kv + (size_t)t * D_LAT;
    float s = dt_bias[h];
    for (int l = 0; l < D_LAT; ++l) s += c[l] * W_dt[l * NH_M + h];
    float sp = (s > 20.f) ? s : log1pf(__expf(s));
    dt[id]  = sp;
    dec[id] = __expf(-__expf(A_log[h]) * sp);
}

// ---------------------------------------------------------------------------
// RoPE in-place on (T, H, 64) buffer
// ---------------------------------------------------------------------------
__global__ __launch_bounds__(256) void rope_kernel(float* __restrict__ buf,
                                                   const float* __restrict__ cosb,
                                                   const float* __restrict__ sinb) {
    int id = blockIdx.x * 256 + threadIdx.x;      // T*H*32
    if (id >= T_LEN * N_HEADS * 32) return;
    int t = id / (N_HEADS * 32);
    int rem = id % (N_HEADS * 32);
    int h = rem >> 5, d = rem & 31;
    size_t base = (size_t)t * D_MODEL + h * DH + d;
    float c0 = cosb[t * DH + d],      s0 = sinb[t * DH + d];
    float c1 = cosb[t * DH + d + 32], s1 = sinb[t * DH + d + 32];
    float a = buf[base], b = buf[base + 32];
    buf[base]      = a * c0 - b * s0;
    buf[base + 32] = b * c1 + a * s1;
}

// ---------------------------------------------------------------------------
// Mamba-3 scan: one workgroup, full SSM state (8*64*128 f32 = 256 KB) lives in LDS.
// 1024 threads: thread owns (h,d)=tid>>1 and state-half (tid&1)*64.
// ---------------------------------------------------------------------------
__global__ __launch_bounds__(1024) void mamba_scan_kernel(const float* __restrict__ c_kv,
                                                          const float* __restrict__ dt,
                                                          const float* __restrict__ dec,
                                                          const float* __restrict__ Bm,
                                                          const float* __restrict__ Cm,
                                                          const float* __restrict__ D_skip,
                                                          float* __restrict__ states) {
    extern __shared__ float smem[];
    float* st = smem;                 // 8*64*128
    float* Bl = smem + NH_M * HD_M * D_STATE;
    float* Cl = Bl + D_STATE;

    const int tid = threadIdx.x;
    const int hd = tid >> 1;          // 0..511
    const int h = hd >> 6;
    const int d = hd & 63;
    const int s0 = (tid & 1) * 64;
    float* mySt = &st[(size_t)hd * D_STATE + s0];
#pragma unroll 4
    for (int s = 0; s < 64; ++s) mySt[s] = 0.f;
    __syncthreads();

    const float dsk = D_skip[h];
    for (int t = 0; t < T_LEN; ++t) {
        if (tid < D_STATE) Bl[tid] = Bm[(size_t)t * D_STATE + tid];
        else if (tid < 2 * D_STATE) Cl[tid - D_STATE] = Cm[(size_t)t * D_STATE + tid - D_STATE];
        __syncthreads();

        const float x    = c_kv[(size_t)t * D_LAT + h * HD_M + d];
        const float dtv  = dt[t * NH_M + h];
        const float dcay = dec[t * NH_M + h];
        const float coef = dtv * x;
        float ysum = 0.f;
#pragma unroll 8
        for (int s = 0; s < 64; ++s) {
            float v = mySt[s] * dcay + coef * Bl[s0 + s];
            mySt[s] = v;
            ysum += v * Cl[s0 + s];
        }
        ysum += __shfl_xor(ysum, 1, 32);
        if ((tid & 1) == 0)
            states[(size_t)t * D_LAT + h * HD_M + d] = ysum + dsk * x;
        __syncthreads();
    }
}

// ---------------------------------------------------------------------------
// Flash attention (causal) + extra "state" column. One wave per (16 q-rows, head).
// Q@K^T and P@V via WMMA bf16; online softmax via lane shuffles.
// ---------------------------------------------------------------------------
__global__ __launch_bounds__(32) void attn_kernel(const float* __restrict__ q,
                                                  const float* __restrict__ kl,
                                                  const float* __restrict__ vl,
                                                  const float* __restrict__ ks,
                                                  const float* __restrict__ vs,
                                                  float* __restrict__ out) {
    __shared__ __bf16 pLds[16 * 32];       // P tile [qrow][key]
    __shared__ __bf16 vLds[64 * 32];       // V tile transposed [d][key]
    __shared__ float  ssLds[16];

    const int lane = threadIdx.x;
    const int l15 = lane & 15;
    const int loA = (lane < 16) ? 0 : 8;
    const int koB = (lane < 16) ? 0 : 8;
    const int hi8 = (lane < 16) ? 0 : 8;
    const int q0 = blockIdx.x * 16;
    const int h  = blockIdx.y;
    const float scale = 0.125f;   // 1/sqrt(64)

    // Q A-operands (Dh=64 => two K=32 chunks)
    v16bf aQ[2];
    {
        const float* qr = q + (size_t)(q0 + l15) * D_MODEL + h * DH;
        float4 qv[4];
#pragma unroll
        for (int c = 0; c < 2; ++c) {
            qv[0] = *reinterpret_cast<const float4*>(qr + c * 32 + loA);
            qv[1] = *reinterpret_cast<const float4*>(qr + c * 32 + loA + 4);
            qv[2] = *reinterpret_cast<const float4*>(qr + c * 32 + 16 + loA);
            qv[3] = *reinterpret_cast<const float4*>(qr + c * 32 + 16 + loA + 4);
            v16bf a;
#pragma unroll
            for (int u = 0; u < 4; ++u) {
                a[u * 4 + 0] = (__bf16)qv[u].x; a[u * 4 + 1] = (__bf16)qv[u].y;
                a[u * 4 + 2] = (__bf16)qv[u].z; a[u * 4 + 3] = (__bf16)qv[u].w;
            }
            aQ[c] = a;
        }
    }

    v8f O[4];
#pragma unroll
    for (int j = 0; j < 4; ++j) O[j] = zero8();
    float mrow[8], lrow[8];
#pragma unroll
    for (int r = 0; r < 8; ++r) { mrow[r] = -1e30f; lrow[r] = 0.f; }

    for (int j0 = 0; j0 < q0 + 16; j0 += 32) {
        // ---- batched K-tile loads for both 16-key subtiles (16 x b128 in flight)
        float4 kv[2][2][4];
#pragma unroll
        for (int jt = 0; jt < 2; ++jt) {
            const int key = j0 + jt * 16 + l15;
            const float* kr = kl + (size_t)key * D_MODEL + h * DH;
#pragma unroll
            for (int c = 0; c < 2; ++c)
#pragma unroll
                for (int u = 0; u < 4; ++u)
                    kv[jt][c][u] = *reinterpret_cast<const float4*>(kr + c * 32 + koB + (u & 1) * 4 + (u >> 1) * 16);
        }
        // ---- scores
        v8f s[2];
#pragma unroll
        for (int jt = 0; jt < 2; ++jt) {
            v8f acc = zero8();
#pragma unroll
            for (int c = 0; c < 2; ++c) {
                v16bf bK;
#pragma unroll
                for (int u = 0; u < 4; ++u) {
                    bK[u * 4 + 0] = (__bf16)kv[jt][c][u].x; bK[u * 4 + 1] = (__bf16)kv[jt][c][u].y;
                    bK[u * 4 + 2] = (__bf16)kv[jt][c][u].z; bK[u * 4 + 3] = (__bf16)kv[jt][c][u].w;
                }
                acc = __builtin_amdgcn_wmma_f32_16x16x32_bf16(
                    false, aQ[c], false, bK, (short)0, acc, false, false);
            }
            const int key = j0 + jt * 16 + l15;
#pragma unroll
            for (int r = 0; r < 8; ++r) {
                float v = acc[r] * scale;
                int query = q0 + r + hi8;
                if (key > query) v = -1e30f;
                acc[r] = v;
            }
            s[jt] = acc;
        }
        // ---- stage V tile transposed (batch loads first): lane handles key j0+lane
        float4 vv[16];
        {
            const float* vr = vl + (size_t)(j0 + lane) * D_MODEL + h * DH;
#pragma unroll
            for (int d4 = 0; d4 < 16; ++d4)
                vv[d4] = *reinterpret_cast<const float4*>(vr + d4 * 4);
        }
        // ---- online softmax per row r
        float alpha[8];
#pragma unroll
        for (int r = 0; r < 8; ++r) {
            float mx = fmaxf(s[0][r], s[1][r]);
#pragma unroll
            for (int off = 1; off < 16; off <<= 1)
                mx = fmaxf(mx, __shfl_xor(mx, off, 32));
            float mn = fmaxf(mrow[r], mx);
            float a  = __expf(mrow[r] - mn);
            float p0 = __expf(s[0][r] - mn);
            float p1 = __expf(s[1][r] - mn);
            float rs = p0 + p1;
#pragma unroll
            for (int off = 1; off < 16; off <<= 1)
                rs += __shfl_xor(rs, off, 32);
            lrow[r] = lrow[r] * a + rs;
            mrow[r] = mn;
            alpha[r] = a;
            int row = r + hi8;
            pLds[row * 32 + l15]      = (__bf16)p0;
            pLds[row * 32 + 16 + l15] = (__bf16)p1;
        }
#pragma unroll
        for (int j = 0; j < 4; ++j)
#pragma unroll
            for (int r = 0; r < 8; ++r) O[j][r] *= alpha[r];

#pragma unroll
        for (int d4 = 0; d4 < 16; ++d4) {
            vLds[(d4 * 4 + 0) * 32 + lane] = (__bf16)vv[d4].x;
            vLds[(d4 * 4 + 1) * 32 + lane] = (__bf16)vv[d4].y;
            vLds[(d4 * 4 + 2) * 32 + lane] = (__bf16)vv[d4].z;
            vLds[(d4 * 4 + 3) * 32 + lane] = (__bf16)vv[d4].w;
        }

        // ---- P A-operand (16x32)
        v16bf aP;
        {
            int row = l15;
            v8bf lo = *reinterpret_cast<const v8bf*>(&pLds[row * 32 + loA]);
            v8bf hi = *reinterpret_cast<const v8bf*>(&pLds[row * 32 + 16 + loA]);
            aP = mk16(lo, hi);
        }
        // ---- P@V: 4 N-tiles of 16
#pragma unroll
        for (int j = 0; j < 4; ++j) {
            int n = j * 16 + l15;
            v8bf lo = *reinterpret_cast<const v8bf*>(&vLds[n * 32 + koB]);
            v8bf hi = *reinterpret_cast<const v8bf*>(&vLds[n * 32 + koB + 16]);
            v16bf bV = mk16(lo, hi);
            O[j] = __builtin_amdgcn_wmma_f32_16x16x32_bf16(
                false, aP, false, bV, (short)0, O[j], false, false);
        }
    }

    // ---- fold in state column: score = q·k_state per row
    if (lane < 16) {
        const float* qr  = q  + (size_t)(q0 + lane) * D_MODEL + h * DH;
        const float* ksr = ks + (size_t)(q0 + lane) * D_MODEL + h * DH;
        float dsum = 0.f;
#pragma unroll 8
        for (int d = 0; d < DH; ++d) dsum += qr[d] * ksr[d];
        ssLds[lane] = dsum * scale;
    }
#pragma unroll
    for (int r = 0; r < 8; ++r) {
        int row = r + hi8;
        float sst = ssLds[row];
        float mf = fmaxf(mrow[r], sst);
        float a  = __expf(mrow[r] - mf);
        float ws = __expf(sst - mf);
        float lf = lrow[r] * a + ws;
        float inv = 1.0f / lf;
        const float* vsr = vs + (size_t)(q0 + row) * D_MODEL + h * DH;
        float* orow = out + (size_t)(q0 + row) * D_MODEL + h * DH;
#pragma unroll
        for (int j = 0; j < 4; ++j) {
            int n = j * 16 + l15;
            float v = O[j][r] * a + ws * vsr[n];
            orow[n] = v * inv;
        }
    }
}

// ---------------------------------------------------------------------------
// fused silu(a1) * a3 -> a1
// ---------------------------------------------------------------------------
__global__ __launch_bounds__(256) void silumul_kernel(float* __restrict__ a1,
                                                      const float* __restrict__ a3, int n) {
    int id = blockIdx.x * 256 + threadIdx.x;
    if (id >= n) return;
    float a = a1[id];
    a1[id] = a / (1.0f + __expf(-a)) * a3[id];
}

// ---------------------------------------------------------------------------
// host launcher
// ---------------------------------------------------------------------------
extern "C" void kernel_launch(void* const* d_in, const int* in_sizes, int n_in,
                              void* d_out, int out_size, void* d_ws, size_t ws_size,
                              hipStream_t stream) {
    (void)in_sizes; (void)n_in; (void)out_size; (void)ws_size;
    const float* x        = (const float*)d_in[0];
    const float* rope_cos = (const float*)d_in[1];
    const float* rope_sin = (const float*)d_in[2];
    const float* norm1_w  = (const float*)d_in[3];
    const float* norm2_w  = (const float*)d_in[4];
    const float* w_q      = (const float*)d_in[5];
    const float* w_kvd    = (const float*)d_in[6];
    const float* w_k_up   = (const float*)d_in[7];
    const float* w_v_up   = (const float*)d_in[8];
    const float* w_st_k   = (const float*)d_in[9];
    const float* w_st_v   = (const float*)d_in[10];
    const float* w_o      = (const float*)d_in[11];
    const float* A_log    = (const float*)d_in[12];
    const float* W_B      = (const float*)d_in[13];
    const float* W_C      = (const float*)d_in[14];
    const float* W_dt     = (const float*)d_in[15];
    const float* dt_bias  = (const float*)d_in[16];
    const float* D_skip   = (const float*)d_in[17];
    const float* mlp_w1   = (const float*)d_in[18];
    const float* mlp_w3   = (const float*)d_in[19];
    const float* mlp_w2   = (const float*)d_in[20];
    float* outp = (float*)d_out;

    float* ws = (float*)d_ws;
    size_t off = 0;
    const size_t TD = (size_t)T_LEN * D_MODEL;
    const size_t TL = (size_t)T_LEN * D_LAT;
    const size_t TF = (size_t)T_LEN * D_FF;
    float* h      = ws + off; off += TD;
    float* q      = ws + off; off += TD;
    float* c_kv   = ws + off; off += TL;
    float* k_loc  = ws + off; off += TD;
    float* v_loc  = ws + off; off += TD;
    float* k_st   = ws + off; off += TD;
    float* v_st   = ws + off; off += TD;
    float* attn   = ws + off; off += TD;
    float* x1     = ws + off; off += TD;
    float* h2     = ws + off; off += TD;
    float* states = ws + off; off += TL;
    float* dt     = ws + off; off += (size_t)T_LEN * NH_M;
    float* dec    = ws + off; off += (size_t)T_LEN * NH_M;
    float* Bm     = ws + off; off += (size_t)T_LEN * D_STATE;
    float* Cm     = ws + off; off += (size_t)T_LEN * D_STATE;
    float* a1     = ws + off; off += TF;
    float* a3     = ws + off; off += TF;

    dim3 blk(256);

    // 1. h = rmsnorm(x)
    rmsnorm_kernel<<<T_LEN, blk, 0, stream>>>(x, norm1_w, h, D_MODEL);
    // 2. q = h @ w_q
    gemm_bf16_kernel<<<dim3(D_MODEL / 128, T_LEN / 128), blk, 0, stream>>>(h, w_q, nullptr, q, T_LEN, D_MODEL, D_MODEL);
    // 3. c_kv = h @ w_kv_down
    gemm_bf16_kernel<<<dim3(D_LAT / 128, T_LEN / 128), blk, 0, stream>>>(h, w_kvd, nullptr, c_kv, T_LEN, D_LAT, D_MODEL);
    // 4/5. k_local / v_local
    gemm_bf16_kernel<<<dim3(D_MODEL / 128, T_LEN / 128), blk, 0, stream>>>(c_kv, w_k_up, nullptr, k_loc, T_LEN, D_MODEL, D_LAT);
    gemm_bf16_kernel<<<dim3(D_MODEL / 128, T_LEN / 128), blk, 0, stream>>>(c_kv, w_v_up, nullptr, v_loc, T_LEN, D_MODEL, D_LAT);
    // 6. Bm, Cm
    gemm_bf16_kernel<<<dim3(1, T_LEN / 128), blk, 0, stream>>>(c_kv, W_B, nullptr, Bm, T_LEN, D_STATE, D_LAT);
    gemm_bf16_kernel<<<dim3(1, T_LEN / 128), blk, 0, stream>>>(c_kv, W_C, nullptr, Cm, T_LEN, D_STATE, D_LAT);
    // 7. dt / decay
    dt_kernel<<<(T_LEN * NH_M + 255) / 256, blk, 0, stream>>>(c_kv, W_dt, dt_bias, A_log, dt, dec);
    // 8. RoPE on q and k_local (in place)
    {
        int n = T_LEN * N_HEADS * 32;
        rope_kernel<<<(n + 255) / 256, blk, 0, stream>>>(q, rope_cos, rope_sin);
        rope_kernel<<<(n + 255) / 256, blk, 0, stream>>>(k_loc, rope_cos, rope_sin);
    }
    // 9. Mamba scan (state fully resident in 256KB LDS)
    {
        size_t smem = (size_t)(NH_M * HD_M * D_STATE + 2 * D_STATE) * sizeof(float);
        mamba_scan_kernel<<<1, 1024, smem, stream>>>(c_kv, dt, dec, Bm, Cm, D_skip, states);
    }
    // 10. k_state / v_state
    gemm_bf16_kernel<<<dim3(D_MODEL / 128, T_LEN / 128), blk, 0, stream>>>(states, w_st_k, nullptr, k_st, T_LEN, D_MODEL, D_LAT);
    gemm_bf16_kernel<<<dim3(D_MODEL / 128, T_LEN / 128), blk, 0, stream>>>(states, w_st_v, nullptr, v_st, T_LEN, D_MODEL, D_LAT);
    // 11. attention
    attn_kernel<<<dim3(T_LEN / 16, N_HEADS), dim3(32), 0, stream>>>(q, k_loc, v_loc, k_st, v_st, attn);
    // 12. x1 = x + attn @ w_o
    gemm_bf16_kernel<<<dim3(D_MODEL / 128, T_LEN / 128), blk, 0, stream>>>(attn, w_o, x, x1, T_LEN, D_MODEL, D_MODEL);
    // 13. h2 = rmsnorm(x1)
    rmsnorm_kernel<<<T_LEN, blk, 0, stream>>>(x1, norm2_w, h2, D_MODEL);
    // 14. a1 = h2 @ w1 ; a3 = h2 @ w3
    gemm_bf16_kernel<<<dim3(D_FF / 128, T_LEN / 128), blk, 0, stream>>>(h2, mlp_w1, nullptr, a1, T_LEN, D_FF, D_MODEL);
    gemm_bf16_kernel<<<dim3(D_FF / 128, T_LEN / 128), blk, 0, stream>>>(h2, mlp_w3, nullptr, a3, T_LEN, D_FF, D_MODEL);
    // 15. a1 = silu(a1) * a3
    silumul_kernel<<<(int)((TF + 255) / 256), blk, 0, stream>>>(a1, a3, (int)TF);
    // 16. out = x1 + a1 @ w2
    gemm_bf16_kernel<<<dim3(D_MODEL / 128, T_LEN / 128), blk, 0, stream>>>(a1, mlp_w2, x1, outp, T_LEN, D_MODEL, D_FF);
}
